// MultiHeadAttention_5042291605886
// MI455X (gfx1250) — compile-verified, matrix operand order checked by reading, not run on hardware
//
#include <hip/hip_runtime.h>
#include <hip/hip_bf16.h>

// ---------------------------------------------------------------------------
// MI455X (gfx1250) fused multi-head attention, bf16 WMMA + async-LDS path.
//   qkv   = x @ Wqkv + b     (WMMA bf16 GEMM; B tile staged via async->LDS)
//   attn  = flash(S^T = K Q^T ; out^T = V^T P^T), K/V tiles async->LDS,
//           double-buffered against the WMMAs (s_wait_asynccnt pipelining)
//   out   = attn @ Wout + b  (WMMA bf16 GEMM, fp32 result)
// ---------------------------------------------------------------------------

typedef __attribute__((ext_vector_type(16))) __bf16 v16bf;
typedef __attribute__((ext_vector_type(8)))  float  v8f;
typedef __attribute__((ext_vector_type(4)))  unsigned v4u;

union FragAB {
  v16bf v;
  v4u   q[2];
  unsigned u[8];
};

__device__ __forceinline__ unsigned short f2bf(float f) {
  unsigned u = __float_as_uint(f);
  u += 0x7FFFu + ((u >> 16) & 1u);   // round-to-nearest-even
  return (unsigned short)(u >> 16);
}

// Load a 16x32 bf16 A/B fragment from a row-major source (global or LDS).
// p = row pointer (row = lane&15) already offset by K0. hf = lane>>4.
// Per ISA 7.12.2: lo 8 elems at K = 8*hf, hi 8 elems at K = 16 + 8*hf.
__device__ __forceinline__ v16bf load_frag(const unsigned short* p, int hf) {
  FragAB f;
  f.q[0] = *(const v4u*)(p + 8 * hf);
  f.q[1] = *(const v4u*)(p + 16 + 8 * hf);
  return f.v;
}

__device__ __forceinline__ v8f wmma_bf16(v16bf a, v16bf b, v8f c) {
  return __builtin_amdgcn_wmma_f32_16x16x32_bf16(false, a, false, b,
                                                 (short)0, c, false, false);
}

// CDNA5 async copy: 16 bytes per lane, global -> LDS, tracked by ASYNCcnt.
// lds_off is the 32-bit LDS byte address (low bits of the flat shared ptr).
__device__ __forceinline__ unsigned lds_addr_of(const void* p) {
  return (unsigned)(unsigned long long)p;
}
__device__ __forceinline__ void async_copy16(unsigned lds_off, const void* g) {
  asm volatile("global_load_async_to_lds_b128 %0, %1, off"
               :: "v"(lds_off), "v"(g)
               : "memory");
}
#define WAIT_ASYNC(n) asm volatile("s_wait_asynccnt " #n ::: "memory")

// ---------------------------------------------------------------------------
// Elementwise fp32 -> bf16
__global__ void cvt_bf16_kernel(const float* __restrict__ in,
                                unsigned short* __restrict__ out, int n) {
  int i = blockIdx.x * 256 + threadIdx.x;
  if (i < n) out[i] = f2bf(in[i]);
}

// Transpose fp32 [K][N] -> bf16 [N][K]
__global__ void transpose_bf16_kernel(const float* __restrict__ in,
                                      unsigned short* __restrict__ out,
                                      int K, int N) {
  size_t idx = (size_t)blockIdx.x * 256 + threadIdx.x;  // over N*K
  int k = (int)(idx % K);
  int n = (int)(idx / K);
  out[idx] = f2bf(in[(size_t)k * N + n]);
}

// ---------------------------------------------------------------------------
// QKV GEMM: D[m,n] = sum_k xb[m,k] * WqkvT[n,k] + bias[n]
// block = 256 (8 waves); wave -> 16(M) x 64(N); all waves share the 64x32
// B tile, staged double-buffered in LDS via async copies. A fragment is
// per-wave, register double-buffered from global.
__global__ void gemm_qkv_kernel(const unsigned short* __restrict__ xb,
                                const unsigned short* __restrict__ wT,
                                const float* __restrict__ bias,
                                unsigned short* __restrict__ Qb,
                                unsigned short* __restrict__ Kb,
                                unsigned short* __restrict__ Vt) {
  __shared__ __align__(16) unsigned short sB[2][64 * 32];
  const int tid = threadIdx.x;
  const int lane = tid & 31;
  const int wave = tid >> 5;
  const int ln = lane & 15, hf = lane >> 4;
  const int mbase = (blockIdx.y * 8 + wave) * 16;
  const int nbase = blockIdx.x * 64;

  const unsigned short* arow = xb + (size_t)(mbase + ln) * 1024;

  auto prefB = [&](int k0, int buf) {
    // 256 threads x 16B = 4KB tile: row r = tid>>2, 16B sub-chunk w = tid&3
    const int r = tid >> 2, w = tid & 3;
    async_copy16(lds_addr_of(&sB[buf][0]) + tid * 16,
                 wT + (size_t)(nbase + r) * 1024 + k0 + w * 8);
  };

  v8f acc[4];
  v8f zero = {};
#pragma unroll
  for (int et = 0; et < 4; ++et) acc[et] = zero;

  v16bf af = load_frag(arow, hf);
  prefB(0, 0);
  for (int t = 0; t < 32; ++t) {
    const int cur = t & 1;
    v16bf af_n = af;
    if (t + 1 < 32) {
      prefB(t * 32 + 32, cur ^ 1);            // async: flies during WMMAs
      af_n = load_frag(arow + t * 32 + 32, hf);
      WAIT_ASYNC(0x1);                        // only tile t must be resident
    } else {
      WAIT_ASYNC(0x0);
    }
    __syncthreads();
#pragma unroll
    for (int et = 0; et < 4; ++et) {
      v16bf bf_ = load_frag(&sB[cur][(et * 16 + ln) * 32], hf);
      acc[et] = wmma_bf16(af, bf_, acc[et]);
    }
    __syncthreads();
    af = af_n;
  }

  const int region = nbase >> 10;          // 0=Q, 1=K, 2=V (tile never spans)
  const int h = (nbase & 1023) >> 6;       // head (64-aligned tile => const)
#pragma unroll
  for (int et = 0; et < 4; ++et) {
    const int n = nbase + et * 16 + ln;
    const float bv = bias[n];
    const int e = n & 63;                  // within-head channel
#pragma unroll
    for (int i = 0; i < 8; ++i) {
      const int m = mbase + hf * 8 + i;    // C layout: row = 8*hf + i
      const int bq = m >> 11;              // batch
      const int nn = m & 2047;             // seq position
      unsigned short v = f2bf(acc[et][i] + bv);
      if (region == 0)
        Qb[((size_t)(bq * 16 + h) * 2048 + nn) * 64 + e] = v;
      else if (region == 1)
        Kb[((size_t)(bq * 16 + h) * 2048 + nn) * 64 + e] = v;
      else
        Vt[((size_t)(bq * 16 + h) * 64 + e) * 2048 + nn] = v;
    }
  }
}

// ---------------------------------------------------------------------------
// Fused flash attention per (b, h). Block = 128 (4 waves), each wave owns a
// 16-query tile; 2048 keys in tiles of 32. K (32x64) and V^T (64x32) tiles
// are shared by all 4 waves: staged in LDS by async copies, double-buffered.
//   S^T = K * Q^T      (4 WMMA)     online softmax (1 shfl_xor per reduce)
//   out^T += V^T * P^T (4 WMMA)     P^T packed straight from S^T C-layout
__global__ void attn_kernel(const unsigned short* __restrict__ Qb,
                            const unsigned short* __restrict__ Kb,
                            const unsigned short* __restrict__ Vt,
                            unsigned short* __restrict__ aout) {
  __shared__ __align__(16) unsigned short sK[2][32 * 64];
  __shared__ __align__(16) unsigned short sV[2][64 * 32];
  const int tid = threadIdx.x;
  const int lane = tid & 31;
  const int wave = tid >> 5;
  const int ln = lane & 15, hf = lane >> 4;
  const int b = blockIdx.z, h = blockIdx.y;
  const int bh = b * 16 + h;
  const int qbase = (blockIdx.x * 4 + wave) * 16;

  // Q^T B-fragments (lane = query column, reads Q row-major), kept in VGPRs
  const unsigned short* qrow = Qb + ((size_t)bh * 2048 + qbase + ln) * 64;
  const v16bf qb0 = load_frag(qrow, hf);        // e 0..31
  const v16bf qb1 = load_frag(qrow + 32, hf);   // e 32..63

  const unsigned short* kbp = Kb + (size_t)bh * 2048 * 64;
  const unsigned short* vbp = Vt + (size_t)bh * 64 * 2048;

  auto prefetch = [&](int kt, int buf) {
    const unsigned sk = lds_addr_of(&sK[buf][0]);
    const unsigned sv = lds_addr_of(&sV[buf][0]);
#pragma unroll
    for (int j = 0; j < 2; ++j) {
      const int c = tid + j * 128;              // 256 x 16B chunks per tile
      // K tile rows are contiguous in global: one 4KB linear block
      async_copy16(sk + c * 16, kbp + (size_t)kt * 64 + c * 8);
      // V^T tile: 64 rows of 64B (stride 4KB in global)
      const int r = c >> 2, w = c & 3;
      async_copy16(sv + c * 16, vbp + (size_t)r * 2048 + kt + w * 8);
    }
  };

  v8f acc[4];
  v8f zero = {};
#pragma unroll
  for (int et = 0; et < 4; ++et) acc[et] = zero;
  float m_run = -1e30f, l_run = 0.f;

  prefetch(0, 0);
  for (int t = 0; t < 64; ++t) {
    const int cur = t & 1;
    if (t + 1 < 64) {
      prefetch((t + 1) * 32, cur ^ 1);          // 4 async insts per wave
      WAIT_ASYNC(0x4);                          // tile t landed; t+1 in flight
    } else {
      WAIT_ASYNC(0x0);
    }
    __syncthreads();

    const unsigned short* k0p = &sK[cur][ln * 64];
    const unsigned short* k1p = &sK[cur][(16 + ln) * 64];
    v16bf ka0 = load_frag(k0p, hf);
    v16bf ka1 = load_frag(k0p + 32, hf);
    v16bf ka2 = load_frag(k1p, hf);
    v16bf ka3 = load_frag(k1p + 32, hf);

    v8f s0 = zero, s1 = zero;
    s0 = wmma_bf16(ka0, qb0, s0);
    s0 = wmma_bf16(ka1, qb1, s0);   // S^T rows = keys kt..kt+15
    s1 = wmma_bf16(ka2, qb0, s1);
    s1 = wmma_bf16(ka3, qb1, s1);   // S^T rows = keys kt+16..kt+31

    // ---- online softmax over keys (per query = per lane column) ----
    float mt = s0[0];
#pragma unroll
    for (int i = 1; i < 8; ++i) mt = fmaxf(mt, s0[i]);
#pragma unroll
    for (int i = 0; i < 8; ++i) mt = fmaxf(mt, s1[i]);
    mt = fmaxf(mt, __shfl_xor(mt, 16, 32));       // combine key halves
    const float m_new = fmaxf(m_run, mt);
    const float alpha = __expf(m_run - m_new);

    float p0[8], p1[8], ts = 0.f;
#pragma unroll
    for (int i = 0; i < 8; ++i) { p0[i] = __expf(s0[i] - m_new); ts += p0[i]; }
#pragma unroll
    for (int i = 0; i < 8; ++i) { p1[i] = __expf(s1[i] - m_new); ts += p1[i]; }
    ts += __shfl_xor(ts, 16, 32);
    l_run = l_run * alpha + ts;
    m_run = m_new;

    // Pack P^T: C-layout of S^T == B-fragment key pattern. No shuffles.
    FragAB pf;
#pragma unroll
    for (int j = 0; j < 4; ++j) {
      pf.u[j]     = (unsigned)f2bf(p0[2 * j]) | ((unsigned)f2bf(p0[2 * j + 1]) << 16);
      pf.u[4 + j] = (unsigned)f2bf(p1[2 * j]) | ((unsigned)f2bf(p1[2 * j + 1]) << 16);
    }

#pragma unroll
    for (int et = 0; et < 4; ++et) {
#pragma unroll
      for (int i = 0; i < 8; ++i) acc[et][i] *= alpha;
    }

    // out^T += V^T * P^T  (A = V^T tile from LDS, row = e-channel)
#pragma unroll
    for (int et = 0; et < 4; ++et) {
      v16bf va = load_frag(&sV[cur][(et * 16 + ln) * 32], hf);
      acc[et] = wmma_bf16(va, pf.v, acc[et]);
    }
    __syncthreads();
  }

  // normalize: softmax sum + reference's post-softmax /sqrt(1024)
  const float scale = 1.0f / (l_run * 32.0f);
  const int q = qbase + ln;
#pragma unroll
  for (int et = 0; et < 4; ++et) {
    union { v4u q4; unsigned short s[8]; } ot;
#pragma unroll
    for (int i = 0; i < 8; ++i) ot.s[i] = f2bf(acc[et][i] * scale);
    unsigned short* dst =
        aout + ((size_t)(b * 2048 + q)) * 1024 + h * 64 + et * 16 + hf * 8;
    *(v4u*)dst = ot.q4;
  }
}

// ---------------------------------------------------------------------------
// Output projection: out[m,n] = sum_k aout[m,k] * WoutT[n,k] + bias[n], fp32.
// Same async-LDS B-tile staging as the QKV GEMM.
__global__ void gemm_out_kernel(const unsigned short* __restrict__ ab,
                                const unsigned short* __restrict__ wT,
                                const float* __restrict__ bias,
                                float* __restrict__ out) {
  __shared__ __align__(16) unsigned short sB[2][64 * 32];
  const int tid = threadIdx.x;
  const int lane = tid & 31;
  const int wave = tid >> 5;
  const int ln = lane & 15, hf = lane >> 4;
  const int mbase = (blockIdx.y * 8 + wave) * 16;
  const int nbase = blockIdx.x * 64;

  const unsigned short* arow = ab + (size_t)(mbase + ln) * 1024;

  auto prefB = [&](int k0, int buf) {
    const int r = tid >> 2, w = tid & 3;
    async_copy16(lds_addr_of(&sB[buf][0]) + tid * 16,
                 wT + (size_t)(nbase + r) * 1024 + k0 + w * 8);
  };

  v8f acc[4];
  v8f zero = {};
#pragma unroll
  for (int et = 0; et < 4; ++et) acc[et] = zero;

  v16bf af = load_frag(arow, hf);
  prefB(0, 0);
  for (int t = 0; t < 32; ++t) {
    const int cur = t & 1;
    v16bf af_n = af;
    if (t + 1 < 32) {
      prefB(t * 32 + 32, cur ^ 1);
      af_n = load_frag(arow + t * 32 + 32, hf);
      WAIT_ASYNC(0x1);
    } else {
      WAIT_ASYNC(0x0);
    }
    __syncthreads();
#pragma unroll
    for (int et = 0; et < 4; ++et) {
      v16bf bf_ = load_frag(&sB[cur][(et * 16 + ln) * 32], hf);
      acc[et] = wmma_bf16(af, bf_, acc[et]);
    }
    __syncthreads();
    af = af_n;
  }

#pragma unroll
  for (int et = 0; et < 4; ++et) {
    const int n = nbase + et * 16 + ln;
    const float bv = bias[n];
#pragma unroll
    for (int i = 0; i < 8; ++i) {
      const int m = mbase + hf * 8 + i;
      out[(size_t)m * 1024 + n] = acc[et][i] + bv;
    }
  }
}

// ---------------------------------------------------------------------------
extern "C" void kernel_launch(void* const* d_in, const int* in_sizes, int n_in,
                              void* d_out, int out_size, void* d_ws, size_t ws_size,
                              hipStream_t stream) {
  (void)in_sizes; (void)n_in; (void)out_size; (void)ws_size;
  const float* x    = (const float*)d_in[0];   // [2,2048,1024]
  const float* Wqkv = (const float*)d_in[1];   // [1024,3072]
  const float* bqkv = (const float*)d_in[2];   // [3072]
  const float* Wout = (const float*)d_in[3];   // [1024,1024]
  const float* bout = (const float*)d_in[4];   // [1024]
  float* out = (float*)d_out;

  // workspace partition (bf16 = unsigned short), ~48 MB total
  unsigned short* xb    = (unsigned short*)d_ws;                  // 4096x1024
  unsigned short* wqkvT = xb    + (size_t)4096 * 1024;            // 3072x1024
  unsigned short* woutT = wqkvT + (size_t)3072 * 1024;            // 1024x1024
  unsigned short* Qb    = woutT + (size_t)1024 * 1024;            // [2][16][2048][64]
  unsigned short* Kb    = Qb    + (size_t)2 * 16 * 2048 * 64;
  unsigned short* Vt    = Kb    + (size_t)2 * 16 * 2048 * 64;     // [2][16][64][2048]
  unsigned short* aout  = Vt    + (size_t)2 * 16 * 2048 * 64;     // 4096x1024

  // 1) precision staging: bf16 copies / transposed bf16 weights
  cvt_bf16_kernel<<<16384, 256, 0, stream>>>(x, xb, 4096 * 1024);
  transpose_bf16_kernel<<<12288, 256, 0, stream>>>(Wqkv, wqkvT, 1024, 3072);
  transpose_bf16_kernel<<<4096, 256, 0, stream>>>(Wout, woutT, 1024, 1024);

  // 2) QKV projection (M=4096, N=3072, K=1024)
  gemm_qkv_kernel<<<dim3(48, 32), 256, 0, stream>>>(xb, wqkvT, bqkv, Qb, Kb, Vt);

  // 3) fused flash attention: 32 q-tiles x 16 heads x 2 batches
  attn_kernel<<<dim3(32, 16, 2), 128, 0, stream>>>(Qb, Kb, Vt, aout);

  // 4) output projection (M=4096, N=1024, K=1024), fp32 result + bias
  gemm_out_kernel<<<dim3(16, 32), 256, 0, stream>>>(aout, woutT, bout, out);
}
